// HyperLoRALinear_13365938225192
// MI455X (gfx1250) — compile-verified, matrix operand order checked by reading, not run on hardware
//
#include <hip/hip_runtime.h>
#include <hip/hip_bf16.h>

// Problem constants (match reference).
#define B_    8
#define S_    4096
#define DIN_  1024
#define DOUT_ 1024
#define R_    16
#define NSK_  16

typedef __attribute__((ext_vector_type(16))) __bf16 v16bf;
typedef __attribute__((ext_vector_type(8)))  __bf16 v8bf;
typedef __attribute__((ext_vector_type(8)))  float  v8f;

// ---------------------------------------------------------------------------
// CDNA5 async global->LDS copy (ASYNCcnt path). Inline asm per bridge doc:
// portable across ROCm7.2 / amdgpu-toolchain. VDST = per-lane LDS byte addr.
// ---------------------------------------------------------------------------
__device__ __forceinline__ void async_load_b128_to_lds(unsigned lds_off, const void* gaddr) {
    asm volatile("global_load_async_to_lds_b128 %0, %1, off"
                 :: "v"(lds_off), "v"(gaddr)
                 : "memory");
}
__device__ __forceinline__ void wait_asynccnt0() {
#if __has_builtin(__builtin_amdgcn_s_wait_asynccnt)
    __builtin_amdgcn_s_wait_asynccnt(0);
#else
    asm volatile("s_wait_asynccnt 0x0" ::: "memory");
#endif
}

// ---------------------------------------------------------------------------
// Kernel 1: task embedding -> 2-layer MLP -> te[B][NSK]   (trivial, 1 block)
// ---------------------------------------------------------------------------
__global__ void te_kernel(const float* __restrict__ emb,
                          const int*   __restrict__ task_ids,
                          const float* __restrict__ w1, const float* __restrict__ b1,
                          const float* __restrict__ w2, const float* __restrict__ b2,
                          float* __restrict__ te_out) {
    __shared__ float t0[B_][NSK_];
    __shared__ float h [B_][NSK_];
    int tid = threadIdx.x;          // 128 threads
    int b = tid >> 4, j = tid & 15;
    t0[b][j] = emb[task_ids[b] * NSK_ + j];
    __syncthreads();
    float s = b1[j];
    #pragma unroll
    for (int k = 0; k < NSK_; ++k) s += t0[b][k] * w1[j * NSK_ + k];
    h[b][j] = s > 0.0f ? s : 0.0f;
    __syncthreads();
    float s2 = b2[j];
    #pragma unroll
    for (int k = 0; k < NSK_; ++k) s2 += h[b][k] * w2[j * NSK_ + k];
    te_out[tid] = s2;
}

// ---------------------------------------------------------------------------
// Kernel 2: A[b, DIN*R] = te[b] . hyper_wA rows ; Bm[b, R*DOUT] likewise.
// ---------------------------------------------------------------------------
__global__ void ab_kernel(const float* __restrict__ te,
                          const float* __restrict__ wA,
                          const float* __restrict__ wB,
                          float* __restrict__ A, float* __restrict__ Bm) {
    int idx  = blockIdx.x * blockDim.x + threadIdx.x;  // < 262144
    int half = idx >> 17;                              // 0 -> A, 1 -> Bm
    int i    = idx & 131071;
    int b    = i >> 14;                                // / (DIN*R) == 16384
    int m    = i & 16383;
    const float* w = (half ? wB : wA) + m * NSK_;
    const float* t = te + b * NSK_;
    float s = 0.0f;
    #pragma unroll
    for (int k = 0; k < NSK_; ++k) s += t[k] * w[k];
    (half ? Bm : A)[i] = s;
}

// ---------------------------------------------------------------------------
// Kernel 3: fold LoRA into effective weight (bf16), layout [b][n][k]:
//   W_eff[b,n,k] = base_w[n,k] + (1/R) * sum_r A[b,k,r] * Bm[b,r,n]
// ---------------------------------------------------------------------------
__global__ void weff_kernel(const float* __restrict__ base_w,
                            const float* __restrict__ A,
                            const float* __restrict__ Bm,
                            __bf16* __restrict__ W) {
    size_t idx = (size_t)blockIdx.x * blockDim.x + threadIdx.x;  // < 8M
    int k = (int)(idx & 1023);
    int n = (int)((idx >> 10) & 1023);
    int b = (int)(idx >> 20);
    const float* a  = A  + b * (DIN_ * R_) + k * R_;   // contiguous 16
    const float* bm = Bm + b * (R_ * DOUT_) + n;       // stride DOUT_
    float s = 0.0f;
    #pragma unroll
    for (int r = 0; r < R_; ++r) s += a[r] * bm[r * DOUT_];
    float val = base_w[n * DIN_ + k] + s * (1.0f / (float)R_);
    W[idx] = (__bf16)val;
}

// ---------------------------------------------------------------------------
// Kernel 4: main GEMM.  out[b,s,:] = x[b,s,:] @ W_eff[b]^T + bias
// 128x128 block tile, BK=32, 8 wave32s (4M x 2N wave grid), 2-deep software
// pipeline: double-buffered LDS, async global->LDS for the bf16 W tile,
// VGPR-staged fp32->bf16 conversion for the x tile. One barrier per K-step.
// ---------------------------------------------------------------------------
#define BM 128
#define BN 128
#define BK 32
#define LDK 40   // padded LDS row stride in halves (80 B: 16B-aligned, spreads banks)

__global__ __launch_bounds__(256, 2)
void gemm_kernel(const float*  __restrict__ x,
                 const __bf16* __restrict__ weff,
                 const float*  __restrict__ bias,
                 float*        __restrict__ out) {
    __shared__ __bf16 sA[2][BM * LDK];   // x tile,  [m][k], bf16
    __shared__ __bf16 sB[2][BN * LDK];   // W tile,  [n][k], bf16

    const int b   = blockIdx.z;
    const int m0  = blockIdx.x * BM;
    const int n0  = blockIdx.y * BN;
    const int tid = threadIdx.x;
    const int lane = tid & 31;
    const int wid  = tid >> 5;
    const int wm   = wid & 3;   // wave M index: 4 waves -> 32 rows each
    const int wn   = wid >> 2;  // wave N index: 2 waves -> 64 cols each
    const int l    = lane & 15;
    const int hiL  = (lane >> 4) & 1;   // half-wave selects K sub-block

    const float*  xb = x    + ((size_t)b * S_ + m0) * DIN_;
    const __bf16* wb = weff + ((size_t)b * DOUT_ + n0) * DIN_;

    // x-tile loader mapping: 8 threads/row (4 floats each), 32 rows/pass, 4 passes
    const int xrow = tid >> 3;
    const int xcol = (tid & 7) * 4;
    // W-tile async mapping: two 16B chunks per thread (rows 0..63 and 64..127)
    const int wrow = tid >> 2;          // 0..63
    const int wkg  = (tid & 3) * 8;     // 0,8,16,24

    v8f acc[2][4];
    const v8f vzero = {0.f, 0.f, 0.f, 0.f, 0.f, 0.f, 0.f, 0.f};
    #pragma unroll
    for (int i = 0; i < 2; ++i)
        #pragma unroll
        for (int j = 0; j < 4; ++j) acc[i][j] = vzero;

    float4 xr[4];

    // ---- prologue: stage tile 0 into buffer 0 ----
    {
        // async W tile (bf16 global -> LDS, no VGPR round trip)
        async_load_b128_to_lds(
            (unsigned)(size_t)&sB[0][wrow * LDK + wkg],
            wb + (size_t)wrow * DIN_ + wkg);
        async_load_b128_to_lds(
            (unsigned)(size_t)&sB[0][(wrow + 64) * LDK + wkg],
            wb + (size_t)(wrow + 64) * DIN_ + wkg);
        #pragma unroll
        for (int p = 0; p < 4; ++p)
            xr[p] = *(const float4*)(xb + (size_t)(xrow + p * 32) * DIN_ + xcol);
        #pragma unroll
        for (int p = 0; p < 4; ++p) {
            __bf16* d = &sA[0][(xrow + p * 32) * LDK + xcol];
            d[0] = (__bf16)xr[p].x; d[1] = (__bf16)xr[p].y;
            d[2] = (__bf16)xr[p].z; d[3] = (__bf16)xr[p].w;
        }
        wait_asynccnt0();
    }
    __syncthreads();

    int cur = 0;
    for (int kk = 0; kk < DIN_; kk += BK) {
        const int nxt = cur ^ 1;
        const bool more = (kk + BK) < DIN_;

        // ---- issue next tile's global traffic before computing this one ----
        if (more) {
            const int kn = kk + BK;
            async_load_b128_to_lds(
                (unsigned)(size_t)&sB[nxt][wrow * LDK + wkg],
                wb + (size_t)wrow * DIN_ + kn + wkg);
            async_load_b128_to_lds(
                (unsigned)(size_t)&sB[nxt][(wrow + 64) * LDK + wkg],
                wb + (size_t)(wrow + 64) * DIN_ + kn + wkg);
            #pragma unroll
            for (int p = 0; p < 4; ++p)
                xr[p] = *(const float4*)(xb + (size_t)(xrow + p * 32) * DIN_ + kn + xcol);
            if (kn + BK < DIN_)
                __builtin_prefetch(xb + (size_t)xrow * DIN_ + kn + BK, 0, 1);
        }

        // ---- load fragments per ISA 16-bit A/B layout, compute 8 WMMAs ----
        // lanes 0-15: K 0..7 + 16..23; lanes 16-31: K 8..15 + 24..31
        v16bf afrag[2], bfrag[4];
        #pragma unroll
        for (int i = 0; i < 2; ++i) {
            const __bf16* p = &sA[cur][(wm * 32 + i * 16 + l) * LDK + hiL * 8];
            v8bf lo = *(const v8bf*)p;
            v8bf hi = *(const v8bf*)(p + 16);
            afrag[i] = __builtin_shufflevector(lo, hi, 0,1,2,3,4,5,6,7,8,9,10,11,12,13,14,15);
        }
        #pragma unroll
        for (int j = 0; j < 4; ++j) {
            const __bf16* p = &sB[cur][(wn * 64 + j * 16 + l) * LDK + hiL * 8];
            v8bf lo = *(const v8bf*)p;
            v8bf hi = *(const v8bf*)(p + 16);
            bfrag[j] = __builtin_shufflevector(lo, hi, 0,1,2,3,4,5,6,7,8,9,10,11,12,13,14,15);
        }
        #pragma unroll
        for (int i = 0; i < 2; ++i)
            #pragma unroll
            for (int j = 0; j < 4; ++j)
                acc[i][j] = __builtin_amdgcn_wmma_f32_16x16x32_bf16(
                    false, afrag[i], false, bfrag[j],
                    (short)0, acc[i][j], false, false);

        // ---- finish staging next tile ----
        if (more) {
            #pragma unroll
            for (int p = 0; p < 4; ++p) {
                __bf16* d = &sA[nxt][(xrow + p * 32) * LDK + xcol];
                d[0] = (__bf16)xr[p].x; d[1] = (__bf16)xr[p].y;
                d[2] = (__bf16)xr[p].z; d[3] = (__bf16)xr[p].w;
            }
            wait_asynccnt0();    // drain async LDS writes before handing off
        }
        __syncthreads();
        cur = nxt;
    }

    // ---- epilogue: C layout is VGPR v -> M = v + 8*hiL, N = lane&15 ----
    const int mbase = m0 + wm * 32;
    const int nbase = n0 + wn * 64;
    #pragma unroll
    for (int j = 0; j < 4; ++j) {
        int n = nbase + j * 16 + l;
        float bv = bias[n];
        #pragma unroll
        for (int i = 0; i < 2; ++i) {
            int mrow0 = mbase + i * 16 + hiL * 8;
            float* op = out + ((size_t)b * S_ + mrow0) * DOUT_ + n;
            #pragma unroll
            for (int v = 0; v < 8; ++v)
                op[(size_t)v * DOUT_] = acc[i][j][v] + bv;
        }
    }
}

// ---------------------------------------------------------------------------
extern "C" void kernel_launch(void* const* d_in, const int* in_sizes, int n_in,
                              void* d_out, int out_size, void* d_ws, size_t ws_size,
                              hipStream_t stream) {
    const float* x       = (const float*)d_in[0];
    const int*   task_id = (const int*)  d_in[1];
    const float* emb     = (const float*)d_in[2];
    const float* w1      = (const float*)d_in[3];
    const float* b1      = (const float*)d_in[4];
    const float* w2      = (const float*)d_in[5];
    const float* b2      = (const float*)d_in[6];
    const float* hwA     = (const float*)d_in[7];
    const float* hwB     = (const float*)d_in[8];
    const float* base_w  = (const float*)d_in[9];
    const float* base_b  = (const float*)d_in[10];
    float* out = (float*)d_out;

    // Workspace layout (≈17.1 MB total):
    //   [0, 1 KB)        te   : B*NSK f32
    //   [1 KB, +512 KB)  A    : B*DIN*R f32
    //   [..., +512 KB)   Bm   : B*R*DOUT f32
    //   [..., +16 MB)    Weff : B*DOUT*DIN bf16
    char*   ws = (char*)d_ws;
    float*  te = (float*)(ws);
    float*  A  = (float*)(ws + 1024);
    float*  Bm = (float*)(ws + 1024 + 524288);
    __bf16* W  = (__bf16*)(ws + 1024 + 2 * 524288);

    te_kernel  <<<1,     128, 0, stream>>>(emb, task_id, w1, b1, w2, b2, te);
    ab_kernel  <<<1024,  256, 0, stream>>>(te, hwA, hwB, A, Bm);
    weff_kernel<<<32768, 256, 0, stream>>>(base_w, A, Bm, W);

    dim3 grid(S_ / BM, DOUT_ / BN, B_);   // 32 x 8 x 8 blocks
    gemm_kernel<<<grid, 256, 0, stream>>>(x, W, base_b, out);
}
